// Encoder_54339926229533
// MI455X (gfx1250) — matrix operand order repaired and checked by reading, not compiled, not run on hardware
//
#include <hip/hip_runtime.h>
#include <hip/hip_bf16.h>
#include <stddef.h>
#include <stdint.h>

// GRU encoder: VOCAB=32000, EMB=256, UNITS=1024, B=64, T=128
#define VOCAB 32000
#define EMB   256
#define UNITS 1024
#define NB    64
#define NT    128
#define THREE (3 * UNITS)   // 3072
#define KSPLIT 4            // K-quarters per gate in the step kernel

typedef __attribute__((ext_vector_type(16))) __bf16 v16bf;
typedef __attribute__((ext_vector_type(8)))  __bf16 v8bf;
typedef __attribute__((ext_vector_type(4)))  __bf16 v4bf;
typedef __attribute__((ext_vector_type(8)))  float  v8f;
typedef __attribute__((ext_vector_type(4)))  float  v4f;

static __device__ __forceinline__ v16bf pack2(v8bf lo, v8bf hi) {
    v16bf r;
#pragma unroll
    for (int i = 0; i < 8; ++i) { r[i] = lo[i]; r[8 + i] = hi[i]; }
    return r;
}

static __device__ __forceinline__ float sigmoidf_(float x) {
    return 1.0f / (1.0f + __expf(-x));
}

// ---------------------------------------------------------------------------
// Prep: transpose + f32 -> bf16 convert.  dst[c*R + r] = src[r*C + c]
// ---------------------------------------------------------------------------
__global__ void conv_transpose_bf16(const float* __restrict__ src,
                                    __bf16* __restrict__ dst, int R, int C) {
    size_t i = (size_t)blockIdx.x * blockDim.x + threadIdx.x;
    if (i >= (size_t)R * C) return;
    int r = (int)(i / C);
    int c = (int)(i % C);
    dst[(size_t)c * R + r] = (__bf16)src[i];
}

__global__ void init_h(const float* __restrict__ h0,
                       __bf16* __restrict__ hbf, float* __restrict__ hf, int n) {
    int i = blockIdx.x * blockDim.x + threadIdx.x;
    if (i >= n) return;
    float v = h0[i];
    hbf[i] = (__bf16)v;
    hf[i]  = v;
}

// ---------------------------------------------------------------------------
// Embedding gather + f32->bf16 convert, done ONCE (not per column tile).
// xemb[r][k] = (bf16) E[x[r]][k],  r = b*T + t.  4 elements per thread.
// ---------------------------------------------------------------------------
__global__ void gather_emb_bf16(const int* __restrict__ x,
                                const float* __restrict__ E,
                                __bf16* __restrict__ xemb) {
    int i = blockIdx.x * blockDim.x + threadIdx.x;      // over (B*T) * EMB / 4
    if (i >= NB * NT * (EMB / 4)) return;
    int r  = i / (EMB / 4);
    int k4 = (i % (EMB / 4)) * 4;
    const float* erow = E + (size_t)x[r] * EMB + k4;
    v4f f = *(const v4f*)erow;
    v4bf o;
#pragma unroll
    for (int j = 0; j < 4; ++j) o[j] = (__bf16)f[j];
    *(v4bf*)(xemb + (size_t)r * EMB + k4) = o;
}

// ---------------------------------------------------------------------------
// Phase 1: xin = xemb @ W + b0   (8192 x 256) x (256 x 3072)
// Each wave owns one 16-row tile and FOUR 16-col tiles: A is loaded once per
// K-chunk and reused for 4 v_wmma_f32_16x16x32_bf16 issues.
// grid = (3072/256, 8192/16) = (12, 512), block = 128 (4 waves)
// ---------------------------------------------------------------------------
__global__ __launch_bounds__(128)
void xin_gemm(const __bf16* __restrict__ xemb, const __bf16* __restrict__ WT,
              const float* __restrict__ b0, float* __restrict__ xin) {
    const int lane   = threadIdx.x & 31;
    const int wave   = threadIdx.x >> 5;
    const int laneHi = lane >> 4;
    const int ln     = lane & 15;

    const int mt   = blockIdx.y;                         // 512 row tiles
    const int ct0  = blockIdx.x * 16 + wave * 4;         // first of 4 col tiles

    const __bf16* arow = xemb + (size_t)(mt * 16 + ln) * EMB;
    const __bf16* wcol[4];
#pragma unroll
    for (int n = 0; n < 4; ++n)
        wcol[n] = WT + (size_t)((ct0 + n) * 16 + ln) * EMB;

    v8f acc[4] = {};
#pragma unroll
    for (int kc = 0; kc < EMB; kc += 32) {
        // A operand: contiguous K order (16 consecutive K per lane half), matches B
        v16bf a = *(const v16bf*)(arow + kc + laneHi * 16);
#pragma unroll
        for (int n = 0; n < 4; ++n) {
            v16bf bm = *(const v16bf*)(wcol[n] + kc + laneHi * 16);
            acc[n] = __builtin_amdgcn_wmma_f32_16x16x32_bf16(
                         false, a, false, bm, (short)0, acc[n], false, false);
        }
    }

#pragma unroll
    for (int n = 0; n < 4; ++n) {
        const int col = (ct0 + n) * 16 + ln;
        const float bias = b0[col];
#pragma unroll
        for (int j = 0; j < 8; ++j) {
            int row = mt * 16 + laneHi * 8 + j;   // D: VGPR j -> M=j (+8 hi lanes)
            xin[(size_t)row * THREE + col] = acc[j == j ? n : n][j] + bias;
        }
    }
}

// ---------------------------------------------------------------------------
// Phase 2 (per time step): rec = h @ U + b1, fused gates, state update.
// Block = one 16-unit column group (ct). 12 waves = 3 gates x 4 K-quarters.
// Each wave computes ALL FOUR m-tiles (M=64) for its gate/K-quarter: the B
// operand is loaded once per chunk and reused by 4 WMMAs, so U traffic is the
// 6 MB/step minimum (L2-resident). Partials reduced through LDS, then the
// sigmoid/tanh gate epilogue runs across the whole block.
// grid = 64 blocks, block = 384 threads (12 waves), 768 waves/step
// ---------------------------------------------------------------------------
__global__ __launch_bounds__(384)
void gru_step(const __bf16* __restrict__ h_src_bf,
              const float*  __restrict__ h_src_f,
              __bf16* __restrict__ h_dst_bf,
              float*  __restrict__ h_dst_f,
              const __bf16* __restrict__ UT,
              const float*  __restrict__ xin,
              const float*  __restrict__ b1,
              float* __restrict__ out,       // [B, T, UNITS]
              float* __restrict__ state_out, // [B, UNITS]
              int t, int is_last) {
    __shared__ float lds[3 * KSPLIT * 4 * 256];   // 48 KB of the 320 KB WGP LDS

    const int lane   = threadIdx.x & 31;
    const int wave   = threadIdx.x >> 5;          // 0..11
    const int laneHi = lane >> 4;
    const int ln     = lane & 15;

    const int gate  = wave >> 2;                  // 0=z 1=r 2=h
    const int kp    = wave & 3;                   // K-quarter
    const int ct    = blockIdx.x;                 // 0..63 unit tile
    const int kbase = kp * (UNITS / KSPLIT);      // 256-wide K slice

    const __bf16* ucol = UT + (size_t)(gate * UNITS + ct * 16 + ln) * UNITS;

    v8f acc[4] = {};
    for (int kc = kbase; kc < kbase + UNITS / KSPLIT; kc += 32) {
        v16bf bm = *(const v16bf*)(ucol + kc + laneHi * 16);
#pragma unroll
        for (int mt = 0; mt < 4; ++mt) {
            const __bf16* hrow = h_src_bf + (size_t)(mt * 16 + ln) * UNITS;
            v16bf a = *(const v16bf*)(hrow + kc + laneHi * 16);
            acc[mt] = __builtin_amdgcn_wmma_f32_16x16x32_bf16(
                          false, a, false, bm, (short)0, acc[mt], false, false);
        }
    }

    // stash partial tiles: slot = wave (gate*4+kp), tile element e = lane*8+j
#pragma unroll
    for (int mt = 0; mt < 4; ++mt) {
#pragma unroll
        for (int j = 0; j < 8; ++j)
            lds[(wave * 4 + mt) * 256 + lane * 8 + j] = acc[mt][j];
    }
    __syncthreads();

    // reduce K-quarters + fused GRU gates over 4*256 = 1024 output elements
    for (int idx = threadIdx.x; idx < 1024; idx += 384) {
        const int mt = idx >> 8;
        const int e  = idx & 255;
        const int l  = e >> 3;               // original lane
        const int j  = e & 7;                // original VGPR index
        const int row = mt * 16 + ((l >> 4) << 3) + j;   // batch index
        const int u   = ct * 16 + (l & 15);              // unit index

        float sz = 0.f, sr = 0.f, sh = 0.f;
#pragma unroll
        for (int q = 0; q < KSPLIT; ++q) {
            sz += lds[((0 * KSPLIT + q) * 4 + mt) * 256 + e];
            sr += lds[((1 * KSPLIT + q) * 4 + mt) * 256 + e];
            sh += lds[((2 * KSPLIT + q) * 4 + mt) * 256 + e];
        }

        const float* xi = xin + ((size_t)row * NT + t) * THREE;
        float z  = sigmoidf_(xi[u]             + sz + b1[u]);
        float r  = sigmoidf_(xi[UNITS + u]     + sr + b1[UNITS + u]);
        float hh = tanhf(xi[2 * UNITS + u] + r * (sh + b1[2 * UNITS + u]));
        float hold = h_src_f[(size_t)row * UNITS + u];
        float hn = z * hold + (1.0f - z) * hh;

        h_dst_f [(size_t)row * UNITS + u] = hn;
        h_dst_bf[(size_t)row * UNITS + u] = (__bf16)hn;
        out[((size_t)row * NT + t) * UNITS + u] = hn;
        if (is_last) state_out[(size_t)row * UNITS + u] = hn;
    }
}

// ---------------------------------------------------------------------------
// Host-side launch
// ---------------------------------------------------------------------------
extern "C" void kernel_launch(void* const* d_in, const int* in_sizes, int n_in,
                              void* d_out, int out_size, void* d_ws, size_t ws_size,
                              hipStream_t stream) {
    (void)in_sizes; (void)n_in; (void)out_size; (void)ws_size;

    const int*   x      = (const int*)  d_in[0];  // [B, T]
    const float* hidden = (const float*)d_in[1];  // [B, UNITS]
    const float* E      = (const float*)d_in[2];  // [VOCAB, EMB]
    const float* W      = (const float*)d_in[3];  // [EMB, 3*UNITS]
    const float* U      = (const float*)d_in[4];  // [UNITS, 3*UNITS]
    const float* b      = (const float*)d_in[5];  // [2, 3*UNITS]
    const float* b0 = b;
    const float* b1 = b + THREE;

    float* out       = (float*)d_out;                              // [B,T,UNITS]
    float* state_out = (float*)d_out + (size_t)NB * NT * UNITS;    // [B,UNITS]

    // Workspace carve-up (256-byte aligned slices); ~108 MB total
    uint8_t* ws = (uint8_t*)d_ws;
    size_t off = 0;
    auto take = [&](size_t bytes) { uint8_t* p = ws + off;
                                    off += (bytes + 255) & ~(size_t)255; return p; };
    __bf16* WT    = (__bf16*)take((size_t)THREE * EMB   * sizeof(__bf16)); // 1.5 MB
    __bf16* UT    = (__bf16*)take((size_t)THREE * UNITS * sizeof(__bf16)); // 6 MB
    __bf16* xemb  = (__bf16*)take((size_t)NB * NT * EMB * sizeof(__bf16)); // 4 MB
    __bf16* hbf0  = (__bf16*)take((size_t)NB * UNITS * sizeof(__bf16));
    __bf16* hbf1  = (__bf16*)take((size_t)NB * UNITS * sizeof(__bf16));
    float*  hf0   = (float*) take((size_t)NB * UNITS * sizeof(float));
    float*  hf1   = (float*) take((size_t)NB * UNITS * sizeof(float));
    float*  xin   = (float*) take((size_t)NB * NT * THREE * sizeof(float)); // 96 MB

    // --- prep: convert/transpose weights, gather embeddings, seed h ---
    {
        int nW = EMB * THREE;
        conv_transpose_bf16<<<(nW + 255) / 256, 256, 0, stream>>>(W, WT, EMB, THREE);
        int nU = UNITS * THREE;
        conv_transpose_bf16<<<(nU + 255) / 256, 256, 0, stream>>>(U, UT, UNITS, THREE);
        int nG = NB * NT * (EMB / 4);
        gather_emb_bf16<<<(nG + 255) / 256, 256, 0, stream>>>(x, E, xemb);
        int nh = NB * UNITS;
        init_h<<<(nh + 255) / 256, 256, 0, stream>>>(hidden, hbf0, hf0, nh);
    }

    // --- phase 1: time-parallel input projection ---
    {
        dim3 grid(THREE / 256, (NB * NT) / 16);   // (12, 512)
        xin_gemm<<<grid, 128, 0, stream>>>(xemb, WT, b0, xin);
    }

    // --- phase 2: sequential recurrence, one fused kernel per step ---
    __bf16* hbf[2] = { hbf0, hbf1 };
    float*  hf [2] = { hf0,  hf1  };
    for (int t = 0; t < NT; ++t) {
        int s = t & 1, d = s ^ 1;
        gru_step<<<64, 384, 0, stream>>>(hbf[s], hf[s], hbf[d], hf[d],
                                         UT, xin, b1, out, state_out,
                                         t, (t == NT - 1) ? 1 : 0);
    }
}
